// NeuralMemory_19499151524354
// MI455X (gfx1250) — matrix operand order repaired and checked by reading, not compiled, not run on hardware
//
#include <hip/hip_runtime.h>
#include <hip/hip_bf16.h>

#define DD    128
#define NN    512
#define NLAY  4

typedef __bf16 bf16_t;
typedef __attribute__((ext_vector_type(16))) __bf16 v16bf;
typedef __attribute__((ext_vector_type(8)))  __bf16 v8bf;
typedef __attribute__((ext_vector_type(8)))  float  v8f;

__device__ __forceinline__ bf16_t f2bf(float f) { return (bf16_t)f; }   // hw cvt, RNE
__device__ __forceinline__ float sigmoidf_(float x) { return 1.f / (1.f + __expf(-x)); }

// ---------------------------------------------------------------------------
// C[M x P] = A[M x 128] @ B[128 x P], bf16 operands, f32 accumulate.
// A row-major bf16; Bt = B^T (P x 128) row-major bf16 -> every lane fragment
// is contiguous: 4 x global_load_b128 + 1 wmma per K-step, K fully unrolled.
// One wave per 16x16 C tile, 4 waves per block.
// ---------------------------------------------------------------------------
__global__ void gemm_bf16(const bf16_t* __restrict__ A,
                          const bf16_t* __restrict__ Bt,
                          float* __restrict__ C, int ldc, int M, int P) {
  const int wave = threadIdx.x >> 5;
  const int lane = threadIdx.x & 31;
  const int tm = blockIdx.x;
  const int tp = blockIdx.y * 4 + wave;
  if (tm * 16 >= M || tp * 16 >= P) return;
  const int half = lane >> 4;        // lane-half (ISA 7.12.2 split)
  const int l16  = lane & 15;        // A row / B,D column
  const bf16_t* arow = A  + (tm * 16 + l16) * DD;
  const bf16_t* brow = Bt + (tp * 16 + l16) * DD;

  v8f acc = {};
#pragma unroll
  for (int k0 = 0; k0 < DD; k0 += 32) {
    union { v16bf v; v8bf h[2]; } a, b;
    a.h[0] = *(const v8bf*)(arow + k0 + 8 * half);
    a.h[1] = *(const v8bf*)(arow + k0 + 16 + 8 * half);
    b.h[0] = *(const v8bf*)(brow + k0 + 8 * half);
    b.h[1] = *(const v8bf*)(brow + k0 + 16 + 8 * half);
    acc = __builtin_amdgcn_wmma_f32_16x16x32_bf16(false, a.v, false, b.v,
                                                  (short)0, acc, false, false);
  }
#pragma unroll
  for (int r = 0; r < 8; ++r)
    C[(tm * 16 + r + 8 * half) * ldc + tp * 16 + l16] = acc[r];
}

// ---------------------------------------------------------------------------
// Conversion / elementwise helpers (bf16 outputs fused into producers)
// ---------------------------------------------------------------------------
__global__ void cvt_rows_kernel(const float* __restrict__ in, int ldi,
                                bf16_t* __restrict__ out) {
  int r = blockIdx.x, c = threadIdx.x;
  out[r * DD + c] = f2bf(in[r * ldi + c]);
}

// transposed convert: out[p*128 + k] = bf16(in[k*ldp + p])   (in is 128 x ldp)
__global__ void cvt_transpose_kernel(const float* __restrict__ in, int ldp,
                                     bf16_t* __restrict__ out) {
  int p = blockIdx.x, k = threadIdx.x;
  out[p * DD + k] = f2bf(in[k * ldp + p]);
}

__global__ void silu_kernel(const float* __restrict__ z, float* __restrict__ a,
                            bf16_t* __restrict__ ab, int n) {
  int i = blockIdx.x * blockDim.x + threadIdx.x;
  if (i < n) { float x = z[i]; float v = x * sigmoidf_(x); a[i] = v; ab[i] = f2bf(v); }
}

// dz = da * silu'(z),  silu'(z) = s*(1 + z*(1-s))
__global__ void dsilu_kernel(const float* __restrict__ da, const float* __restrict__ z,
                             float* __restrict__ dz, bf16_t* __restrict__ dzb, int n) {
  int i = blockIdx.x * blockDim.x + threadIdx.x;
  if (i < n) {
    float zz = z[i], s = sigmoidf_(zz);
    float v = da[i] * (s * (1.f + zz * (1.f - s)));
    dz[i] = v; dzb[i] = f2bf(v);
  }
}

// dz4 = (2/D)*(pred - values); values live in kvbuf[:,128:256]
__global__ void dz4_kernel(const float* __restrict__ pred, const float* __restrict__ kvbuf,
                           float* __restrict__ dz4, bf16_t* __restrict__ dz4b) {
  int i = blockIdx.x, c = threadIdx.x;
  float v = (2.0f / (float)DD) * (pred[i * DD + c] - kvbuf[i * 256 + 128 + c]);
  dz4[i * DD + c] = v; dz4b[i * DD + c] = f2bf(v);
}

// per-token scalars: nlr = -(seq@W_step), am = seq@W_mom, d1m = 1 - sigmoid(seq@W_decay)
__global__ void scalars_kernel(const float* __restrict__ seq,
                               const float* __restrict__ W_step,
                               const float* __restrict__ W_mom,
                               const float* __restrict__ W_decay,
                               float* __restrict__ nlr, float* __restrict__ am,
                               float* __restrict__ d1m) {
  __shared__ float red[3][DD];
  int i = blockIdx.x, t = threadIdx.x;
  float x = seq[i * DD + t];
  red[0][t] = x * W_step[t];
  red[1][t] = x * W_mom[t];
  red[2][t] = x * W_decay[t];
  __syncthreads();
  for (int s = 64; s > 0; s >>= 1) {
    if (t < s) { red[0][t] += red[0][t + s]; red[1][t] += red[1][t + s]; red[2][t] += red[2][t + s]; }
    __syncthreads();
  }
  if (t == 0) {
    nlr[i] = -red[0][0];
    am[i]  =  red[1][0];
    d1m[i] = 1.f - sigmoidf_(red[2][0]);
  }
}

// ---------------------------------------------------------------------------
// Fused sequential scan + per-token retrieval (one persistent WGP workgroup).
//   u[4][128][128] f32 -> 256 KB dynamic LDS (of 320 KB WGP pool)
//   m[4][128][128] f32 -> 64 VGPRs/thread, striped e = tid + 1024*k
//   W_mem slices       -> 64 VGPRs/thread
// Software-pipelined: token i+1's staging loads are issued between the
// recurrence update and retrieval so their L2 latency hides under the
// 4-layer matvec; token i+2 is prefetched (global_prefetch_b8).
// ---------------------------------------------------------------------------
__global__ void __launch_bounds__(1024)
scan_retrieve_kernel(const float* __restrict__ kvbuf,
                     const float* __restrict__ a1, const float* __restrict__ a2,
                     const float* __restrict__ a3,
                     const float* __restrict__ dz1, const float* __restrict__ dz2,
                     const float* __restrict__ dz3, const float* __restrict__ dz4,
                     const float* __restrict__ qbuf,
                     const float* __restrict__ W_mem,
                     const float* __restrict__ nlr, const float* __restrict__ am,
                     const float* __restrict__ d1m,
                     float* __restrict__ out) {
  extern __shared__ float lds[];
  float* u    = lds;                     // 65536 f32
  float* actb = lds + 65536;             // [0..511]=acts, [512..1023]=deltas
  float* hbuf = actb + 1024;             // 128 f32
  float* red  = hbuf + 128;              // 1024 f32

  const int t   = threadIdx.x;
  const int c   = t & 127;
  const int seg = t >> 7;

#pragma unroll
  for (int k = 0; k < 64; ++k) u[t + k * 1024] = 0.f;
  float m[64];
#pragma unroll
  for (int k = 0; k < 64; ++k) m[k] = 0.f;

  float wm[NLAY][16];
#pragma unroll
  for (int l = 0; l < NLAY; ++l)
#pragma unroll
    for (int j = 0; j < 16; ++j)
      wm[l][j] = W_mem[l * DD * DD + (seg * 16 + j) * DD + c];

  // per-thread staging source (vec 0-3: acts, 4-7: deltas), resolved once
  const int vec = t >> 7, r = t & 127;
  const float* srcp; int sstr;
  switch (vec) {
    case 0:  srcp = kvbuf + r; sstr = 256; break;   // act0 = keys
    case 1:  srcp = a1 + r;    sstr = DD;  break;
    case 2:  srcp = a2 + r;    sstr = DD;  break;
    case 3:  srcp = a3 + r;    sstr = DD;  break;
    case 4:  srcp = dz1 + r;   sstr = DD;  break;
    case 5:  srcp = dz2 + r;   sstr = DD;  break;
    case 6:  srcp = dz3 + r;   sstr = DD;  break;
    default: srcp = dz4 + r;   sstr = DD;  break;
  }

  // prime the pipeline with token 0
  float stage = srcp[0];
  float qv    = (t < DD) ? qbuf[t] : 0.f;
  float ami = am[0], di = d1m[0], li = nlr[0];
  __syncthreads();

  for (int i = 0; i < NN; ++i) {
    actb[t] = stage;
    if (t < DD) hbuf[t] = qv;
    __syncthreads();

    // recurrence: s = -lr * act_l[r] * delta_l[c];  m = am*m + s;  u = (1-dec)*u + m
#pragma unroll
    for (int k = 0; k < 64; ++k) {
      int e  = t + k * 1024;
      int l  = e >> 14;
      int rr = (e >> 7) & 127;
      int cc = e & 127;
      float s  = li * actb[l * DD + rr] * actb[512 + l * DD + cc];
      float mm = ami * m[k] + s;
      m[k] = mm;
      u[e] = di * u[e] + mm;
    }

    // issue token i+1's staging loads now; they complete under retrieval
    if (i + 1 < NN) {
      stage = srcp[(i + 1) * sstr];
      if (t < DD) qv = qbuf[(i + 1) * DD + t];
      ami = am[i + 1]; di = d1m[i + 1]; li = nlr[i + 1];
    }
    if (i + 2 < NN) {
      __builtin_prefetch(srcp + (i + 2) * sstr, 0, 3);       // global_prefetch_b8
      if (t < DD) __builtin_prefetch(qbuf + (i + 2) * DD + t, 0, 3);
    }
    __syncthreads();

    // retrieval: h <- q; for l: h = (silu)(h @ (W_mem[l] + u[l]))
#pragma unroll
    for (int l = 0; l < NLAY; ++l) {
      const float* ul = u + l * DD * DD;
      float acc = 0.f;
#pragma unroll
      for (int j = 0; j < 16; ++j) {
        int rr = seg * 16 + j;
        acc += hbuf[rr] * (wm[l][j] + ul[rr * DD + c]);
      }
      red[t] = acc;
      __syncthreads();
      if (t < DD) {
        float o = 0.f;
#pragma unroll
        for (int sgi = 0; sgi < 8; ++sgi) o += red[sgi * DD + t];
        if (l < NLAY - 1) hbuf[t] = o * sigmoidf_(o);
        else              out[i * DD + t] = o;
      }
      __syncthreads();
    }
  }
}

// ---------------------------------------------------------------------------
extern "C" void kernel_launch(void* const* d_in, const int* in_sizes, int n_in,
                              void* d_out, int out_size, void* d_ws, size_t ws_size,
                              hipStream_t stream) {
  (void)in_sizes; (void)n_in; (void)out_size; (void)ws_size;
  const float* seq     = (const float*)d_in[0];
  const float* W_mem   = (const float*)d_in[1];
  const float* W_q     = (const float*)d_in[2];
  const float* W_kv    = (const float*)d_in[3];
  const float* W_mom   = (const float*)d_in[4];
  const float* W_step  = (const float*)d_in[5];
  const float* W_decay = (const float*)d_in[6];
  float* out = (float*)d_out;

  // ---- f32 workspace region ----
  float* ws    = (float*)d_ws;
  float* kvbuf = ws;                    // 512*256
  float* qbuf  = kvbuf + NN * 256;
  float* z1    = qbuf + NN * DD;
  float* a1    = z1   + NN * DD;
  float* z2    = a1   + NN * DD;
  float* a2    = z2   + NN * DD;
  float* z3    = a2   + NN * DD;
  float* a3    = z3   + NN * DD;
  float* z4    = a3   + NN * DD;
  float* dz4   = z4   + NN * DD;
  float* da    = dz4  + NN * DD;
  float* dz3   = da   + NN * DD;
  float* dz2   = dz3  + NN * DD;
  float* dz1   = dz2  + NN * DD;
  float* nlr   = dz1  + NN * DD;        // 512 each
  float* am    = nlr  + NN;
  float* d1m   = am   + NN;

  // ---- bf16 workspace region ----
  bf16_t* bb    = (bf16_t*)(d1m + NN);
  bf16_t* seqb  = bb;                   // 512*128 each below
  bf16_t* keysb = seqb  + NN * DD;
  bf16_t* a1b   = keysb + NN * DD;
  bf16_t* a2b   = a1b   + NN * DD;
  bf16_t* a3b   = a2b   + NN * DD;
  bf16_t* dz4b  = a3b   + NN * DD;
  bf16_t* dz3b  = dz4b  + NN * DD;
  bf16_t* dz2b  = dz3b  + NN * DD;
  bf16_t* dz1b  = dz2b  + NN * DD;
  bf16_t* WqT   = dz1b  + NN * DD;      // 128*128
  bf16_t* WkvT  = WqT   + DD * DD;      // 256*128
  bf16_t* WT    = WkvT  + 2 * DD * DD;  // W0T..W3T : 4*128*128
  bf16_t* Wb    = WT    + 4 * DD * DD;  // W1..W3 row-major bf16 : 3*128*128

  const dim3 gblk(128);
  const int n = NN * DD;

  // one-time operand conversions (weights: Bt layouts; Wb = bf16(W) acts as (W^T)^T)
  cvt_rows_kernel     <<<NN, DD, 0, stream>>>(seq, DD, seqb);
  cvt_transpose_kernel<<<DD, DD, 0, stream>>>(W_q, DD, WqT);
  cvt_transpose_kernel<<<2*DD, DD, 0, stream>>>(W_kv, 2*DD, WkvT);
  for (int l = 0; l < 4; ++l)
    cvt_transpose_kernel<<<DD, DD, 0, stream>>>(W_mem + l*DD*DD, DD, WT + l*DD*DD);
  for (int l = 1; l < 4; ++l)
    cvt_rows_kernel<<<DD, DD, 0, stream>>>(W_mem + l*DD*DD, DD, Wb + (l-1)*DD*DD);

  // token-parallel GEMMs (WMMA bf16, K unrolled -> 4 wmma each)
  gemm_bf16<<<dim3(NN/16, 4), gblk, 0, stream>>>(seqb, WkvT, kvbuf, 2*DD, NN, 2*DD);
  gemm_bf16<<<dim3(NN/16, 2), gblk, 0, stream>>>(seqb, WqT, qbuf, DD, NN, DD);
  cvt_rows_kernel<<<NN, DD, 0, stream>>>(kvbuf, 2*DD, keysb);   // keys -> bf16
  gemm_bf16<<<dim3(NN/16, 2), gblk, 0, stream>>>(keysb, WT + 0*DD*DD, z1, DD, NN, DD);
  silu_kernel<<<n/256, 256, 0, stream>>>(z1, a1, a1b, n);
  gemm_bf16<<<dim3(NN/16, 2), gblk, 0, stream>>>(a1b, WT + 1*DD*DD, z2, DD, NN, DD);
  silu_kernel<<<n/256, 256, 0, stream>>>(z2, a2, a2b, n);
  gemm_bf16<<<dim3(NN/16, 2), gblk, 0, stream>>>(a2b, WT + 2*DD*DD, z3, DD, NN, DD);
  silu_kernel<<<n/256, 256, 0, stream>>>(z3, a3, a3b, n);
  gemm_bf16<<<dim3(NN/16, 2), gblk, 0, stream>>>(a3b, WT + 3*DD*DD, z4, DD, NN, DD);

  // backward deltas:  da_l = dz_{l+1} @ W_l^T  ->  Bt = bf16(W_l) row-major
  dz4_kernel<<<NN, DD, 0, stream>>>(z4, kvbuf, dz4, dz4b);
  gemm_bf16<<<dim3(NN/16, 2), gblk, 0, stream>>>(dz4b, Wb + 2*DD*DD, da, DD, NN, DD);
  dsilu_kernel<<<n/256, 256, 0, stream>>>(da, z3, dz3, dz3b, n);
  gemm_bf16<<<dim3(NN/16, 2), gblk, 0, stream>>>(dz3b, Wb + 1*DD*DD, da, DD, NN, DD);
  dsilu_kernel<<<n/256, 256, 0, stream>>>(da, z2, dz2, dz2b, n);
  gemm_bf16<<<dim3(NN/16, 2), gblk, 0, stream>>>(dz2b, Wb + 0*DD*DD, da, DD, NN, DD);
  dsilu_kernel<<<n/256, 256, 0, stream>>>(da, z1, dz1, dz1b, n);

  scalars_kernel<<<NN, DD, 0, stream>>>(seq, W_step, W_mom, W_decay, nlr, am, d1m);

  // fused scan + retrieval: 256 KB u-state in LDS + staging/reduction buffers
  size_t ldsbytes = (size_t)(65536 + 1024 + 128 + 1024) * sizeof(float);  // 270848 B
  hipFuncSetAttribute((const void*)scan_retrieve_kernel,
                      hipFuncAttributeMaxDynamicSharedMemorySize, (int)ldsbytes);
  scan_retrieve_kernel<<<1, 1024, ldsbytes, stream>>>(
      kvbuf, a1, a2, a3, dz1, dz2, dz3, dz4, qbuf, W_mem, nlr, am, d1m, out);
}
